// CLFormer_11965778887139
// MI455X (gfx1250) — matrix operand order repaired and checked by reading, not compiled, not run on hardware
//
#include <hip/hip_runtime.h>
#include <math.h>

typedef __attribute__((ext_vector_type(16))) _Float16 v16h;
typedef __attribute__((ext_vector_type(8)))  float    v8f;

#define B_     32
#define C_     32
#define L_     16384
#define H_     4
#define DH     8
#define NB     3
#define DOUT_  10
#define CHUNKS 8
#define CHROWS (L_ / CHUNKS)      // 2048
#define TILES  (B_ * L_ / 16)     // 32768 16-token tiles
#define WFRAG_PER_BLK (2 * 2 * 32 * 16)  // gemm x tile x lane x 16 halfs

// Branch-free exact-erf GELU (Abramowitz-Stegun 7.1.26, |err| <= 1.5e-7).
// No exec-mask divergence -> clean WMMA-compatible instruction stream.
__device__ __forceinline__ float gelu_exact(float x) {
    const float z  = x * 0.70710678118654752440f;
    const float az = fabsf(z);
    const float t  = __builtin_amdgcn_rcpf(fmaf(0.3275911f, az, 1.0f));
    float p = fmaf(t, 1.061405429f, -1.453152027f);
    p = fmaf(t, p, 1.421413741f);
    p = fmaf(t, p, -0.284496736f);
    p = fmaf(t, p, 0.254829592f);
    p *= t;
    const float erfz = 1.0f - p * __expf(-az * az);
    return 0.5f * x * (1.0f + copysignf(erfz, x));
}

// Async global -> LDS copy helpers (gfx1250 ASYNCcnt path; bypasses VGPRs).
// ldsoff = low 32 bits of flat shared address == LDS byte offset.
__device__ __forceinline__ unsigned lds_off(const void* p) {
    return (unsigned)(size_t)p;
}
__device__ __forceinline__ void async_g2l_b128(unsigned ldsoff, const void* g) {
    asm volatile("global_load_async_to_lds_b128 %0, %1, off"
                 :: "v"(ldsoff), "v"((unsigned long long)g) : "memory");
}
__device__ __forceinline__ void async_g2l_b32(unsigned ldsoff, const void* g) {
    asm volatile("global_load_async_to_lds_b32 %0, %1, off"
                 :: "v"(ldsoff), "v"((unsigned long long)g) : "memory");
}
__device__ __forceinline__ void wait_async0() {
    asm volatile("s_wait_asynccnt 0x0" ::: "memory");
}

// ---------------------------------------------------------------------------
// 0) x [B,C,L] -> h [B,L,C]; async global->LDS with transposing LDS addressing
//    (one async op per element replaces load+store pair), coalesced writes.
// ---------------------------------------------------------------------------
__global__ void transpose_kernel(const float* __restrict__ x, float* __restrict__ h) {
    __shared__ float t[32][33];
    const int b  = blockIdx.x / (L_ / 32);
    const int l0 = (blockIdx.x % (L_ / 32)) * 32;
    const int lt = threadIdx.x & 31;
    const int g  = threadIdx.x >> 5;             // 8 groups
    const unsigned base = lds_off(&t[0][0]);
    for (int c = g; c < 32; c += 8)
        async_g2l_b32(base + (unsigned)(c * 33 + lt) * 4,
                      x + ((size_t)b * C_ + c) * L_ + l0 + lt);
    wait_async0();
    __syncthreads();
    const int cc = threadIdx.x & 31;
    for (int r = g; r < 32; r += 8)
        h[((size_t)b * L_ + l0 + r) * C_ + cc] = t[cc][r];
}

// ---------------------------------------------------------------------------
// 1) fcW1/fcW2 (f32, [NB][32][32]) -> f16 WMMA B-fragments.
//    Dense 16-bit B (32x16, wave32): lane n<16 holds column n, K=0..7,16..23
//    (2 per VGPR); lane 16+n holds column n, K=8..15,24..31.
// ---------------------------------------------------------------------------
__global__ void wprep_kernel(const float* __restrict__ W1, const float* __restrict__ W2,
                             _Float16* __restrict__ wfrag) {
    int idx = blockIdx.x * blockDim.x + threadIdx.x;
    if (idx >= NB * WFRAG_PER_BLK) return;
    int kk    = idx & 15;
    int lane  = (idx >> 4) & 31;
    int ntile = (idx >> 9) & 1;
    int gemm  = (idx >> 10) & 1;
    int blk   = idx >> 11;
    int col   = ntile * 16 + (lane & 15);
    int k     = ((lane < 16) ? 0 : 8) + ((kk & 8) << 1) + (kk & 7);
    const float* W = (gemm == 0) ? W1 : W2;
    wfrag[idx] = (_Float16)W[((size_t)blk * 32 + k) * 32 + col];
}

// ---------------------------------------------------------------------------
// 2) Per-(b,c) online softmax stats over an L-chunk (max, sumexp)
// ---------------------------------------------------------------------------
__global__ void stats_partial_kernel(const float* __restrict__ h, float* __restrict__ statsP) {
    const int b   = blockIdx.x / CHUNKS;
    const int ch  = blockIdx.x % CHUNKS;
    const int c   = threadIdx.x & 31;
    const int sub = threadIdx.x >> 5;           // 8 sub-reducers per channel
    const float* hb = h + ((size_t)b * L_ + (size_t)ch * CHROWS) * C_;
    float m = -3.4e38f, s = 0.0f;
    for (int r = sub; r < CHROWS; r += 8) {
        __builtin_prefetch(hb + (size_t)(r + 64) * C_ + c, 0, 1);
        float v  = hb[(size_t)r * C_ + c];
        float mn = fmaxf(m, v);
        s = s * __expf(m - mn) + __expf(v - mn);
        m = mn;
    }
    __shared__ float sm[256], ss[256];
    sm[threadIdx.x] = m; ss[threadIdx.x] = s;
    __syncthreads();
    for (int st = 128; st >= 32; st >>= 1) {
        if (threadIdx.x < st) {
            float m2 = sm[threadIdx.x + st], s2 = ss[threadIdx.x + st];
            float mn = fmaxf(sm[threadIdx.x], m2);
            ss[threadIdx.x] = ss[threadIdx.x] * __expf(sm[threadIdx.x] - mn) + s2 * __expf(m2 - mn);
            sm[threadIdx.x] = mn;
        }
        __syncthreads();
    }
    if (threadIdx.x < 32) {
        statsP[((size_t)(ch * B_ + b) * C_ + c) * 2 + 0] = sm[threadIdx.x];
        statsP[((size_t)(ch * B_ + b) * C_ + c) * 2 + 1] = ss[threadIdx.x];
    }
}

// ---------------------------------------------------------------------------
// 3) Merge chunk stats -> (max, 1/sum); zero kv accumulators (+pool on flag)
// ---------------------------------------------------------------------------
__global__ void stats_final_kernel(const float* __restrict__ statsP, float* __restrict__ statsF,
                                   float* __restrict__ kv, float* __restrict__ pool, int zero_pool) {
    int idx = blockIdx.x * blockDim.x + threadIdx.x;   // 8192 threads
    if (idx < B_ * C_) {
        float m = -3.4e38f, s = 0.0f;
        for (int ch = 0; ch < CHUNKS; ++ch) {
            float m2 = statsP[((size_t)ch * B_ * C_ + idx) * 2 + 0];
            float s2 = statsP[((size_t)ch * B_ * C_ + idx) * 2 + 1];
            float mn = fmaxf(m, m2);
            s = s * __expf(m - mn) + s2 * __expf(m2 - mn);
            m = mn;
        }
        statsF[idx * 2 + 0] = m;
        statsF[idx * 2 + 1] = 1.0f / s;
        if (zero_pool) pool[idx] = 0.0f;
    }
    if (idx < B_ * H_ * DH * DH) kv[idx] = 0.0f;
}

// ---------------------------------------------------------------------------
// 4) kv[b][head][d][e] = sum_l exp(h[l,d]-max_d)*invZ_d * h[l,e]  (per head)
// ---------------------------------------------------------------------------
__global__ void kv_kernel(const float* __restrict__ h, const float* __restrict__ statsF,
                          float* __restrict__ kv) {
    const int b    = blockIdx.x / CHUNKS;
    const int ch   = blockIdx.x % CHUNKS;
    const int head = threadIdx.x & 3;
    const int grp  = threadIdx.x >> 2;          // 64 row-groups
    float mx[DH], iz[DH];
#pragma unroll
    for (int d = 0; d < DH; ++d) {
        mx[d] = statsF[((size_t)b * C_ + head * DH + d) * 2 + 0];
        iz[d] = statsF[((size_t)b * C_ + head * DH + d) * 2 + 1];
    }
    float acc[DH][DH] = {};
    const float* hb = h + ((size_t)b * L_ + (size_t)ch * CHROWS) * C_;
    for (int r = grp; r < CHROWS; r += 64) {
        __builtin_prefetch(hb + (size_t)(r + 64) * C_ + head * DH, 0, 1);
        const float4* p = (const float4*)(hb + (size_t)r * C_ + head * DH);
        float4 va = p[0], vb = p[1];
        float v[DH] = {va.x, va.y, va.z, va.w, vb.x, vb.y, vb.z, vb.w};
#pragma unroll
        for (int d = 0; d < DH; ++d) {
            float kdv = __expf(v[d] - mx[d]) * iz[d];
#pragma unroll
            for (int e = 0; e < DH; ++e) acc[d][e] = fmaf(kdv, v[e], acc[d][e]);
        }
    }
    __shared__ float red[H_ * DH * DH];          // 256 floats
    red[threadIdx.x] = 0.0f;
    __syncthreads();
#pragma unroll
    for (int d = 0; d < DH; ++d)
#pragma unroll
        for (int e = 0; e < DH; ++e)
            atomicAdd(&red[(head * DH + d) * DH + e], acc[d][e]);
    __syncthreads();
    atomicAdd(&kv[(size_t)b * H_ * DH * DH + threadIdx.x], red[threadIdx.x]);
}

// ---------------------------------------------------------------------------
// 5) Fused per-token pass: attention-out -> WMMA fc1 + GELU -> WMMA fc2 +
//    GELU, in-place on h.  One wave = one 16-token tile.  Tile staged via
//    async global->LDS (ASYNCcnt), fc GEMMs via v_wmma_f32_16x16x32_f16.
// ---------------------------------------------------------------------------
__device__ __forceinline__ v16h buildA(const float* __restrict__ t /*16x32*/, int lane) {
    const int m  = lane & 15;
    const int kb = (lane < 16) ? 0 : 8;
    v16h a;
#pragma unroll
    for (int kk = 0; kk < 8; ++kk) {
        a[kk]     = (_Float16)t[m * C_ + kb + kk];
        a[kk + 8] = (_Float16)t[m * C_ + 16 + kb + kk];
    }
    return a;
}

__global__ void __launch_bounds__(256)
block_fused_kernel(float* __restrict__ h, const float* __restrict__ kv,
                   const _Float16* __restrict__ wfragBlk,
                   const float* __restrict__ bias1, const float* __restrict__ bias2,
                   float* __restrict__ pool, int do_pool) {
    const int lane = threadIdx.x & 31;
    const int w    = threadIdx.x >> 5;
    const int tile = blockIdx.x * 8 + w;
    const int b    = tile / (L_ / 16);
    const int row0 = (tile % (L_ / 16)) * 16;
    float* hrow = h + ((size_t)b * L_ + row0) * C_;     // 512 contiguous floats

    __shared__ float tileS[8][16][C_];                  // 16 KB
    __shared__ float kvS[H_ * DH * DH];                 // 1 KB (same b for all 8 waves)
    if (threadIdx.x < H_ * DH * DH)
        kvS[threadIdx.x] = kv[(size_t)b * H_ * DH * DH + threadIdx.x];

    // --- stage 16x32 tile: async global->LDS, 16B per lane per op ---
    {
        const unsigned base = lds_off(&tileS[w][0][0]);
#pragma unroll
        for (int k2 = 0; k2 < 4; ++k2)
            async_g2l_b128(base + (unsigned)(k2 * 32 + lane) * 16,
                           hrow + (k2 * 32 + lane) * 4);
        wait_async0();
    }
    __syncthreads();

    // --- attention: lane = (token, head-pair); q-softmax (8-wide) + q@kv ---
    {
        const int tt = lane >> 1, hp = lane & 1;        // cols [16*hp, 16*hp+16)
        float out[16];
#pragma unroll
        for (int s = 0; s < 2; ++s) {
            const int head = hp * 2 + s;
            float v[8], m = -3.4e38f;
#pragma unroll
            for (int d = 0; d < 8; ++d) { v[d] = tileS[w][tt][head * 8 + d]; m = fmaxf(m, v[d]); }
            float q[8], qs = 0.0f;
#pragma unroll
            for (int d = 0; d < 8; ++d) { q[d] = __expf(v[d] - m); qs += q[d]; }
            const float inv = 1.0f / qs;
#pragma unroll
            for (int e = 0; e < 8; ++e) {
                float a = 0.0f;
#pragma unroll
                for (int d = 0; d < 8; ++d) a = fmaf(q[d] * inv, kvS[(head * DH + d) * DH + e], a);
                out[s * 8 + e] = a;
            }
        }
        // lane overwrites exactly the cols it alone read
#pragma unroll
        for (int i2 = 0; i2 < 16; ++i2) tileS[w][tt][hp * 16 + i2] = out[i2];
    }
    __syncthreads();

    const v16h* wf = (const v16h*)wfragBlk;   // [(gemm*2+ntile)*32 + lane]
    const int n    = lane & 15;
    const int mrow = (lane >> 4) * 8;

    // --- fc1: D = A(16x32) x W1(32x32) via two 16x16x32 WMMAs ---
    {
        v16h a  = buildA(&tileS[w][0][0], lane);
        v16h b0 = wf[(0 * 2 + 0) * 32 + lane];
        v16h b1 = wf[(0 * 2 + 1) * 32 + lane];
        v8f c0 = {}, c1 = {};
        c0 = __builtin_amdgcn_wmma_f32_16x16x32_f16(false, a, false, b0, (short)0, c0, false, false);
        c1 = __builtin_amdgcn_wmma_f32_16x16x32_f16(false, a, false, b1, (short)0, c1, false, false);
        __syncthreads();                         // everyone done reading tileS
        const float bb0 = bias1[n], bb1 = bias1[16 + n];
#pragma unroll
        for (int r = 0; r < 8; ++r) {
            tileS[w][mrow + r][n]      = gelu_exact(c0[r] + bb0);
            tileS[w][mrow + r][16 + n] = gelu_exact(c1[r] + bb1);
        }
    }
    __syncthreads();

    // --- fc2 ---
    {
        v16h a  = buildA(&tileS[w][0][0], lane);
        v16h b0 = wf[(1 * 2 + 0) * 32 + lane];
        v16h b1 = wf[(1 * 2 + 1) * 32 + lane];
        v8f c0 = {}, c1 = {};
        c0 = __builtin_amdgcn_wmma_f32_16x16x32_f16(false, a, false, b0, (short)0, c0, false, false);
        c1 = __builtin_amdgcn_wmma_f32_16x16x32_f16(false, a, false, b1, (short)0, c1, false, false);
        __syncthreads();
        const float bb0 = bias2[n], bb1 = bias2[16 + n];
#pragma unroll
        for (int r = 0; r < 8; ++r) {
            tileS[w][mrow + r][n]      = gelu_exact(c0[r] + bb0);
            tileS[w][mrow + r][16 + n] = gelu_exact(c1[r] + bb1);
        }
    }
    __syncthreads();

    // --- in-place store, coalesced float4 ---
    {
        const float4* ls = (const float4*)&tileS[w][0][0];
        float4* go = (float4*)hrow;
#pragma unroll
        for (int k2 = 0; k2 < 4; ++k2) go[k2 * 32 + lane] = ls[k2 * 32 + lane];
    }

    // --- pooling accumulation fused into last block's pass ---
    if (do_pool) {
        float sum = 0.0f;
#pragma unroll
        for (int r = 0; r < 16; ++r) sum += tileS[w][r][lane];
        atomicAdd(&pool[(size_t)b * C_ + lane], sum);
    }
}

// ---------------------------------------------------------------------------
// 6) Head: mean-pool -> Wh+bh -> BN(eval) -> GELU -> Wf+bf  (tiny, 1 wave)
// ---------------------------------------------------------------------------
__global__ void head_kernel(const float* __restrict__ pool, const float* __restrict__ Wh,
                            const float* __restrict__ bh, const float* __restrict__ gam,
                            const float* __restrict__ beta, const float* __restrict__ mean,
                            const float* __restrict__ var, const float* __restrict__ Wf,
                            const float* __restrict__ bf, float* __restrict__ out) {
    const int b = threadIdx.x;
    if (b >= B_) return;
    float p[C_], y[C_];
#pragma unroll
    for (int c = 0; c < C_; ++c) p[c] = pool[(size_t)b * C_ + c] * (1.0f / (float)L_);
    for (int c = 0; c < C_; ++c) {
        float a = bh[c];
#pragma unroll
        for (int k = 0; k < C_; ++k) a = fmaf(p[k], Wh[k * C_ + c], a);
        a = (a - mean[c]) * rsqrtf(var[c] + 1e-5f) * gam[c] + beta[c];
        y[c] = gelu_exact(a);
    }
    for (int j = 0; j < DOUT_; ++j) {
        float a = bf[j];
#pragma unroll
        for (int k = 0; k < C_; ++k) a = fmaf(y[k], Wf[k * DOUT_ + j], a);
        out[(size_t)b * DOUT_ + j] = a;
    }
}

// ---------------------------------------------------------------------------
extern "C" void kernel_launch(void* const* d_in, const int* in_sizes, int n_in,
                              void* d_out, int out_size, void* d_ws, size_t ws_size,
                              hipStream_t stream) {
    const float* x     = (const float*)d_in[0];
    const float* fcW1  = (const float*)d_in[1];
    const float* fcb1  = (const float*)d_in[2];
    const float* fcW2  = (const float*)d_in[3];
    const float* fcb2  = (const float*)d_in[4];
    const float* Wh    = (const float*)d_in[5];
    const float* bh    = (const float*)d_in[6];
    const float* gam   = (const float*)d_in[7];
    const float* beta  = (const float*)d_in[8];
    const float* mean  = (const float*)d_in[9];
    const float* var   = (const float*)d_in[10];
    const float* Wf    = (const float*)d_in[11];
    const float* bf    = (const float*)d_in[12];
    float* out = (float*)d_out;

    // workspace carve-up (~67.2 MB)
    float* ws      = (float*)d_ws;
    float* h       = ws;                                    // B*L*C = 16,777,216 f32
    float* statsP  = h + (size_t)B_ * L_ * C_;              // CHUNKS*B*C*2 = 16384
    float* statsF  = statsP + (size_t)CHUNKS * B_ * C_ * 2; // B*C*2 = 2048
    float* kv      = statsF + (size_t)B_ * C_ * 2;          // B*H*64 = 8192
    float* pool    = kv + (size_t)B_ * H_ * DH * DH;        // B*C = 1024
    _Float16* wfrag = (_Float16*)(pool + B_ * C_);          // NB*WFRAG_PER_BLK halfs (32B-aligned)

    transpose_kernel<<<B_ * (L_ / 32), 256, 0, stream>>>(x, h);
    wprep_kernel<<<(NB * WFRAG_PER_BLK + 255) / 256, 256, 0, stream>>>(fcW1, fcW2, wfrag);

    for (int blk = 0; blk < NB; ++blk) {
        const int last = (blk == NB - 1) ? 1 : 0;
        stats_partial_kernel<<<B_ * CHUNKS, 256, 0, stream>>>(h, statsP);
        stats_final_kernel<<<32, 256, 0, stream>>>(statsP, statsF, kv, pool, last);
        kv_kernel<<<B_ * CHUNKS, 256, 0, stream>>>(h, statsF, kv);
        block_fused_kernel<<<TILES / 8, 256, 0, stream>>>(
            h, kv, wfrag + (size_t)blk * WFRAG_PER_BLK,
            fcb1 + blk * C_, fcb2 + blk * C_, pool, last);
    }
    head_kernel<<<1, 32, 0, stream>>>(pool, Wh, bh, gam, beta, mean, var, Wf, bf, out);
}
